// CSS_94489280978
// MI455X (gfx1250) — matrix operand order repaired and checked by reading, not compiled, not run on hardware
//
#include <hip/hip_runtime.h>
#include <hip/hip_bf16.h>

typedef _Float16 half_t;
typedef __attribute__((ext_vector_type(16))) _Float16 v16h;
typedef __attribute__((ext_vector_type(8)))  float    v8f;
typedef __attribute__((ext_vector_type(4)))  float    v4f;

#define B_DIM   8
#define K_DIM   19
#define K_PAD   32
#define C_DIM   512
#define HW      16384

// ---------------------------------------------------------------------------
// Kernel 1: per-(b,k) softmax statistics: row max and 1/sum(exp(x-max)).
// ---------------------------------------------------------------------------
__global__ void __launch_bounds__(256) k_softmax_stats(const float* __restrict__ probs,
                                                       float* __restrict__ stats) {
    const int row = blockIdx.x;                 // 0..151  (b*19 + k)
    const float* p = probs + (size_t)row * HW;
    __shared__ float sm[256];
    const int tid = threadIdx.x;

    float m = -__builtin_inff();
    for (int i = tid; i < HW; i += 256) m = fmaxf(m, p[i]);
    sm[tid] = m; __syncthreads();
    for (int s = 128; s > 0; s >>= 1) {
        if (tid < s) sm[tid] = fmaxf(sm[tid], sm[tid + s]);
        __syncthreads();
    }
    const float mx = sm[0]; __syncthreads();

    float acc = 0.f;
    for (int i = tid; i < HW; i += 256) acc += __expf(p[i] - mx);
    sm[tid] = acc; __syncthreads();
    for (int s = 128; s > 0; s >>= 1) {
        if (tid < s) sm[tid] += sm[tid + s];
        __syncthreads();
    }
    if (tid == 0) {
        stats[row * 2 + 0] = mx;
        stats[row * 2 + 1] = 1.0f / sm[0];
    }
}

// ---------------------------------------------------------------------------
// Kernel 2: attn transposed + padded as f16: attn_t[b][n][kpad=32].
// Coalesced reads per k, contiguous 64B store per thread, zero-padded k>=19.
// 8.4 MB total -> resident in 192 MB L2 for the GEMM.
// ---------------------------------------------------------------------------
__global__ void __launch_bounds__(256) k_attn_t(const float* __restrict__ probs,
                                                const float* __restrict__ stats,
                                                half_t* __restrict__ attn_t) {
    const int bid = blockIdx.x;                 // b*64 + nchunk
    const int b   = bid >> 6;
    const int n   = ((bid & 63) << 8) + threadIdx.x;

    half_t h[K_PAD];
#pragma unroll
    for (int k = 0; k < K_DIM; ++k) {
        const float mx = stats[(b * K_DIM + k) * 2 + 0];   // uniform -> s_load
        const float rs = stats[(b * K_DIM + k) * 2 + 1];
        const float v  = __expf(probs[((size_t)(b * K_DIM + k)) * HW + n] - mx) * rs;
        h[k] = (half_t)v;
    }
#pragma unroll
    for (int k = K_DIM; k < K_PAD; ++k) h[k] = (half_t)0.f;

    v4f* dst = (v4f*)(attn_t + ((size_t)b * HW + n) * K_PAD);
    const v4f* src = (const v4f*)h;
#pragma unroll
    for (int j = 0; j < 4; ++j) dst[j] = src[j];
}

// ---------------------------------------------------------------------------
// Kernel 3: ctx[b,c,k] = sum_n attn[b,k,n]*feats[b,c,n] via v_wmma_f32_16x16x32_f16.
// Explicit 2-stage software pipeline with disjoint register sets so that
//  - a full stage of 4 b128 + 32 16-bit loads is always in flight (hides HBM/L2 latency)
//  - WMMA->VALU hazard slots are filled with the other stage's loads/cvts.
// feats streamed from HBM exactly once -> ~11.5 us bandwidth floor.
// ---------------------------------------------------------------------------
#define LOAD_STAGE(F0, F1, F2, F3, B0, B1, N0)                              \
    do {                                                                    \
        const float* fp_ = fbase + (N0);                                    \
        F0 = *(const v4f*)(fp_ +  0);                                       \
        F1 = *(const v4f*)(fp_ +  4);                                       \
        F2 = *(const v4f*)(fp_ + 16);                                       \
        F3 = *(const v4f*)(fp_ + 20);                                       \
        const half_t* ap_ = abase + (size_t)((N0) + kb) * K_PAD + (lane & 15); \
        _Pragma("unroll")                                                   \
        for (int s_ = 0; s_ < 16; ++s_) {                                   \
            const int kk_ = (s_ < 8) ? s_ : (s_ + 8);                       \
            B0[s_] = ap_[kk_ * K_PAD];                                      \
            B1[s_] = ap_[kk_ * K_PAD + 16];                                 \
        }                                                                   \
    } while (0)

#define CVT_A(A, F0, F1, F2, F3)                                            \
    do {                                                                    \
        _Pragma("unroll")                                                   \
        for (int j_ = 0; j_ < 4; ++j_) {                                    \
            A[j_]      = (_Float16)F0[j_];                                  \
            A[4 + j_]  = (_Float16)F1[j_];                                  \
            A[8 + j_]  = (_Float16)F2[j_];                                  \
            A[12 + j_] = (_Float16)F3[j_];                                  \
        }                                                                   \
    } while (0)

__global__ void __launch_bounds__(256, 2) k_ctx_wmma(const float* __restrict__ feats,
                                                     const half_t* __restrict__ attn_t,
                                                     float* __restrict__ out) {
    const int blk  = blockIdx.x;                // b*32 + ctile
    const int b    = blk >> 5;
    const int c0   = (blk & 31) << 4;
    const int tid  = threadIdx.x;
    const int wid  = tid >> 5;
    const int lane = tid & 31;
    const int seg0 = wid << 11;                 // 2048 n per wave

    // A-operand addressing: row M = lane%16, K-halves split by half-wave
    const int m  = lane & 15;
    const int kb = (lane < 16) ? 0 : 8;
    const float*  fbase = feats + ((size_t)(b * C_DIM + c0 + m)) * HW + kb;
    const half_t* abase = attn_t + ((size_t)b * HW) * K_PAD;

    v8f acc0 = {};
    v8f acc1 = {};

    v4f f0, f1, f2, f3, g0, g1, g2, g3;
    v16h b0, b1, d0, d1;

    LOAD_STAGE(f0, f1, f2, f3, b0, b1, seg0);           // prologue: stage A

    for (int it = 0; it < 32; ++it) {
        const int base = seg0 + (it << 6);              // 64 n per iteration

        // issue stage-B loads, then retire stage A into the matrix cores
        LOAD_STAGE(g0, g1, g2, g3, d0, d1, base + 32);
        v16h a;
        CVT_A(a, f0, f1, f2, f3);
        acc0 = __builtin_amdgcn_wmma_f32_16x16x32_f16(false, a, false, b0,
                                                      (short)0, acc0, false, false);
        acc1 = __builtin_amdgcn_wmma_f32_16x16x32_f16(false, a, false, b1,
                                                      (short)0, acc1, false, false);

        // issue stage-A loads for next iteration (clamped re-read on the last
        // iteration keeps addresses in-bounds), then retire stage B
        const int npre = (it < 31) ? (base + 64) : seg0;
        LOAD_STAGE(f0, f1, f2, f3, b0, b1, npre);
        v16h a2;
        CVT_A(a2, g0, g1, g2, g3);
        acc0 = __builtin_amdgcn_wmma_f32_16x16x32_f16(false, a2, false, d0,
                                                      (short)0, acc0, false, false);
        acc1 = __builtin_amdgcn_wmma_f32_16x16x32_f16(false, a2, false, d1,
                                                      (short)0, acc1, false, false);
    }

    // ---- cross-wave reduction in LDS (8 waves x 512 partials)
    __shared__ float red[8][512];
#pragma unroll
    for (int r = 0; r < 8; ++r) {
        red[wid][r * 32 + lane]       = acc0[r];
        red[wid][256 + r * 32 + lane] = acc1[r];
    }
    __syncthreads();

#pragma unroll
    for (int t = 0; t < 2; ++t) {
        const int slot = t * 256 + tid;
        float s = 0.f;
#pragma unroll
        for (int w = 0; w < 8; ++w) s += red[w][slot];
        const int tile = slot >> 8;
        const int r    = (slot >> 5) & 7;
        const int l    = slot & 31;
        const int c    = c0 + r + ((l < 16) ? 0 : 8);
        const int k    = tile * 16 + (l & 15);
        if (k < K_DIM) out[((size_t)(b * C_DIM + c)) * K_DIM + k] = s;
    }
}

// ---------------------------------------------------------------------------
extern "C" void kernel_launch(void* const* d_in, const int* in_sizes, int n_in,
                              void* d_out, int out_size, void* d_ws, size_t ws_size,
                              hipStream_t stream) {
    const float* feats = (const float*)d_in[0];   // (8,512,128,128) fp32
    const float* probs = (const float*)d_in[1];   // (8,19,128,128)  fp32
    float* out = (float*)d_out;                   // (8,512,19,1)    fp32

    float*  stats  = (float*)d_ws;                            // 152*2 floats
    half_t* attn_t = (half_t*)((char*)d_ws + 4096);           // 8*16384*32 f16

    k_softmax_stats<<<B_DIM * K_DIM, 256, 0, stream>>>(probs, stats);
    k_attn_t<<<B_DIM * (HW / 256), 256, 0, stream>>>(probs, stats, attn_t);
    k_ctx_wmma<<<B_DIM * (C_DIM / 16), 256, 0, stream>>>(feats, attn_t, out);
}
